// LocalVoxelEncoder_63986422776124
// MI455X (gfx1250) — matrix operand order repaired and checked by reading, not compiled, not run on hardware
//
#include <hip/hip_runtime.h>

typedef __attribute__((ext_vector_type(16))) __bf16 v16bf;
typedef __attribute__((ext_vector_type(8)))  float  v8f;

#define RESO 128
#define CDIM 128
#define DG 64
#define PLANE_ELEMS (2 * CDIM * RESO * RESO)  // per-plane floats (B=2)

// coord = linspace(-0.5,0.5,64)[idx]; v = coord/1.101 + 0.5; clip; floor(v*128).
// Spacing ~1.85 bins/step -> injective per axis -> every occupied cell has count 64.
__device__ __forceinline__ int bin_of(int idx) {
    float coord = -0.5f + (float)idx * (1.0f / 63.0f);
    float v = coord * (1.0f / 1.101f) + 0.5f;
    v = fminf(fmaxf(v, 0.0f), 1.0f - 1e-5f);
    return (int)(v * 128.0f);
}

__device__ __forceinline__ float xval(const float* __restrict__ xb,
                                      int gi, int gj, int gk) {
    if ((unsigned)gi < 64u && (unsigned)gj < 64u && (unsigned)gk < 64u)
        return xb[gi * 4096 + gj * 64 + gk];
    return 0.0f;
}

__global__ void zero_kernel(float4* __restrict__ p, int n4) {
    int i = blockIdx.x * blockDim.x + threadIdx.x;
    if (i < n4) p[i] = make_float4(0.f, 0.f, 0.f, 0.f);
}

// AXIS==0: block (b, i, ktile), loop j -> xz reduction in registers (+ xy partials)
// AXIS==1: block (b, j, ktile), loop i -> yz reduction in registers
// 128 threads = 4 wave32s. All waves share the M-tile (16 k's); wave w owns
// channel tiles nt = 2w, 2w+1 (B fragments held in VGPRs for all 64 iterations).
template <int AXIS>
__global__ __launch_bounds__(128)
void conv_reduce_kernel(const float* __restrict__ x,
                        const float* __restrict__ w,
                        const float* __restrict__ bias,
                        float* __restrict__ out) {
    __shared__ alignas(32) __bf16 AFrag[32][16];  // per-lane WMMA layout, 1 KB
    __shared__ float xysum[CDIM];

    const int tid = threadIdx.x;
    const int blk = blockIdx.x;        // b*256 + fix*4 + kt
    const int kt  = blk & 3;
    const int fix = (blk >> 2) & 63;
    const int b   = blk >> 8;
    const int k0  = kt * 16;

    const int wave = tid >> 5;
    const int lane = tid & 31;
    const int half = lane >> 4;
    const int l16  = lane & 15;

    const float* xb = x + (size_t)b * (DG * DG * DG);

    // B fragments (32x16 KxN tile: lane = n + 16*half; e -> K = 16*(e>>3) + 8*half + (e&7)),
    // built straight from global weights (L2-resident), loop-invariant in VGPRs.
    v16bf bfr[2];
    float bia[2];
#pragma unroll
    for (int t = 0; t < 2; ++t) {
        const int ncol = (wave * 2 + t) * 16 + l16;
        const int kb = half * 16;
#pragma unroll
        for (int e = 0; e < 16; ++e) {
            int K = ((e >> 3) << 4) + half * 8 + (e & 7);
            bfr[t][e] = (__bf16)((K < 27) ? w[ncol * 27 + K] : 0.0f);
        }
        bia[t] = bias[ncol];
    }

    float red_acc[2][8];  // running sum over the loop axis, per (nt, r)
#pragma unroll
    for (int t = 0; t < 2; ++t)
#pragma unroll
        for (int r = 0; r < 8; ++r) red_acc[t][r] = 0.0f;

    for (int lv = 0; lv < DG; ++lv) {
        const int i = (AXIS == 0) ? fix : lv;
        const int j = (AXIS == 0) ? lv : fix;

        __syncthreads();  // previous iteration's AFrag reads are done
        // im2col A fragment for M-tile k0..k0+15 (16-bit A 16x32 layout:
        // lanes 0-15: e0-7->K0-7, e8-15->K16-23; lanes 16-31: +8)
        for (int idx = tid; idx < 32 * 16; idx += 128) {
            int e  = idx & 15;
            int ln = idx >> 4;
            int m  = k0 + (ln & 15);
            int K  = ((e >> 3) << 4) + ((ln & 16) ? 8 : 0) + (e & 7);
            float v = 0.0f;
            if (K < 27) {
                int di = K / 9, rem = K % 9;
                int dj = rem / 3, dk = rem % 3;
                v = xval(xb, i + di - 1, j + dj - 1, m + dk - 1);
            }
            AFrag[ln][e] = (__bf16)v;
        }
        if (AXIS == 0) xysum[tid] = 0.0f;
        __syncthreads();

        const v16bf afrag = *reinterpret_cast<const v16bf*>(&AFrag[lane][0]);

#pragma unroll
        for (int t = 0; t < 2; ++t) {
            v8f c = {0.f, 0.f, 0.f, 0.f, 0.f, 0.f, 0.f, 0.f};
            v8f d = __builtin_amdgcn_wmma_f32_16x16x32_bf16(
                false, afrag, false, bfr[t], (short)0, c, false, false);
            float ksum = 0.0f;
#pragma unroll
            for (int r = 0; r < 8; ++r) {
                float val = fmaxf(d[r] + bia[t], 0.0f) * (1.0f / 64.0f);
                red_acc[t][r] += val;  // xz/yz: reduce over loop axis in registers
                ksum += val;           // xy: reduce over k tile
            }
            if (AXIS == 0)
                atomicAdd(&xysum[wave * 32 + t * 16 + l16], ksum);
        }

        if (AXIS == 0) {
            __syncthreads();
            // one global atomic per (j, ch): 4 contributions per xy cell total
            int bi = bin_of(i), bj = bin_of(j);
            atomicAdd(&out[PLANE_ELEMS + (((b * CDIM + tid) * RESO + bj) * RESO) + bi],
                      xysum[tid]);
        }
    }

    // completed reduction -> direct stores (injective bins: block owns its cells)
    const int bfix = bin_of(fix);
    float* plane = (AXIS == 0) ? out : out + 2 * PLANE_ELEMS;
#pragma unroll
    for (int t = 0; t < 2; ++t) {
        const int ch = wave * 32 + t * 16 + l16;
#pragma unroll
        for (int r = 0; r < 8; ++r) {
            int k = k0 + r + half * 8;   // C/D layout: M = r + 8*half
            int bk = bin_of(k);
            plane[(((b * CDIM + ch) * RESO + bk) * RESO) + bfix] = red_acc[t][r];
        }
    }
}

extern "C" void kernel_launch(void* const* d_in, const int* in_sizes, int n_in,
                              void* d_out, int out_size, void* d_ws, size_t ws_size,
                              hipStream_t stream) {
    const float* x  = (const float*)d_in[0];
    const float* cw = (const float*)d_in[1];
    const float* cb = (const float*)d_in[2];
    float* out = (float*)d_out;

    int n4 = out_size / 4;  // 12,582,912 floats
    zero_kernel<<<(n4 + 255) / 256, 256, 0, stream>>>((float4*)out, n4);
    conv_reduce_kernel<0><<<2 * 64 * 4, 128, 0, stream>>>(x, cw, cb, out);  // xz + xy
    conv_reduce_kernel<1><<<2 * 64 * 4, 128, 0, stream>>>(x, cw, cb, out);  // yz
}
